// MultiHeadAttention_84335977824725
// MI455X (gfx1250) — compile-verified, matrix operand order checked by reading, not run on hardware
//
#include <hip/hip_runtime.h>
#include <hip/hip_bf16.h>

#define B_ 2
#define S_ 2048
#define HID_ 1024
#define NH_ 16
#define D_ 64
#define LN_EPS 1e-12f

typedef __attribute__((ext_vector_type(16))) _Float16 v16h;
typedef __attribute__((ext_vector_type(8)))  _Float16 h8;
typedef __attribute__((ext_vector_type(8)))  float    v8f;

// ---------------------------------------------------------------------------
// WMMA fragment layouts (CDNA5 ISA 7.12.2, wave32):
//   A 16x32 f16:  lane m = lane&15, kbase = (lane&16)?8:0, k-runs
//                 [kbase,kbase+8) and [16+kbase,16+kbase+8)
//   B 32x16 f16:  lane n = lane&15, same k mapping
//   C/D 16x16 f32: vgpr i -> row = i + ((lane&16)?8:0), col = lane&15
// Each k-run is 8 contiguous f16 = one 16B (b128) load.
// ---------------------------------------------------------------------------
__device__ __forceinline__ v16h load_frag(const _Float16* base, int kbase) {
    h8 lo = *(const h8*)(base + kbase);
    h8 hi = *(const h8*)(base + 16 + kbase);
    return __builtin_shufflevector(lo, hi, 0, 1, 2, 3, 4, 5, 6, 7,
                                   8, 9, 10, 11, 12, 13, 14, 15);
}

// ===========================================================================
// Kernel 0: prep pass.  z=0..2: q/k/v f32 -> f16.  z=3..6: W f32[K,N] ->
//           W^T f16[N,K] so GEMM B-fragments become contiguous 16B runs.
// ===========================================================================
__global__ __launch_bounds__(256)
void prep_kernel(const float* __restrict__ q, const float* __restrict__ k,
                 const float* __restrict__ v,
                 const float* __restrict__ wq, const float* __restrict__ wk,
                 const float* __restrict__ wv, const float* __restrict__ wd,
                 _Float16* __restrict__ xh, _Float16* __restrict__ wt) {
    const int z = blockIdx.z;
    const size_t idx = (size_t)blockIdx.x * 256 + threadIdx.x;
    if (z < 3) {
        const float* src = (z == 0) ? q : (z == 1) ? k : v;
        xh[(size_t)z * (B_ * S_ * HID_) + idx] = (_Float16)src[idx];
    } else if (idx < (size_t)HID_ * HID_) {
        const float* W = (z == 3) ? wq : (z == 4) ? wk : (z == 5) ? wv : wd;
        const int kk = (int)(idx >> 10), n = (int)(idx & 1023);
        wt[(size_t)(z - 3) * (HID_ * HID_) + (size_t)n * HID_ + kk] = (_Float16)W[idx];
    }
}

// ===========================================================================
// Software-pipelined 16x64-per-wave GEMM body, FULLY UNROLLED over K:
// no loop-carried phis -> every b128 load result feeds its WMMA directly and
// the register allocator never needs rotation copies.  The per-stage
// sched_barrier(0)s act as pressure fences: each stage's refill loads stay
// below the WMMAs that consume the stage (bounded liveness) while still
// overlapping the neighboring stage's WMMAs.
// ===========================================================================
__device__ __forceinline__ void gemm_16x64(const _Float16* Arow,
                                           const _Float16* W0, const _Float16* W1,
                                           const _Float16* W2, const _Float16* W3,
                                           int kbase, v8f acc[4]) {
    const _Float16* Wrow[4] = {W0, W1, W2, W3};
    v16h a0 = load_frag(Arow, kbase);
    v16h b0[4];
#pragma unroll
    for (int t = 0; t < 4; ++t) b0[t] = load_frag(Wrow[t], kbase);
    v16h a1 = load_frag(Arow + 32, kbase);
    v16h b1[4];
#pragma unroll
    for (int t = 0; t < 4; ++t) b1[t] = load_frag(Wrow[t] + 32, kbase);

#pragma unroll
    for (int kk = 0; kk < HID_ - 64; kk += 64) {
        // ---- consume stage 0 ----
#pragma unroll
        for (int t = 0; t < 4; ++t)
            acc[t] = __builtin_amdgcn_wmma_f32_16x16x32_f16(false, a0, false, b0[t],
                                                            (short)0, acc[t], false, false);
        __builtin_amdgcn_sched_barrier(0);   // refill must stay below the WMMAs
        // ---- refill stage 0 for kk+64 (overlaps stage-1 WMMAs below) ----
        a0 = load_frag(Arow + kk + 64, kbase);
#pragma unroll
        for (int t = 0; t < 4; ++t) b0[t] = load_frag(Wrow[t] + kk + 64, kbase);
        // ---- consume stage 1 ----
#pragma unroll
        for (int t = 0; t < 4; ++t)
            acc[t] = __builtin_amdgcn_wmma_f32_16x16x32_f16(false, a1, false, b1[t],
                                                            (short)0, acc[t], false, false);
        __builtin_amdgcn_sched_barrier(0);   // refill must stay below the WMMAs
        // ---- refill stage 1 for kk+96 (overlaps next step's stage-0 WMMAs) ----
        a1 = load_frag(Arow + kk + 96, kbase);
#pragma unroll
        for (int t = 0; t < 4; ++t) b1[t] = load_frag(Wrow[t] + kk + 96, kbase);
    }
#pragma unroll
    for (int t = 0; t < 4; ++t)
        acc[t] = __builtin_amdgcn_wmma_f32_16x16x32_f16(false, a0, false, b0[t],
                                                        (short)0, acc[t], false, false);
#pragma unroll
    for (int t = 0; t < 4; ++t)
        acc[t] = __builtin_amdgcn_wmma_f32_16x16x32_f16(false, a1, false, b1[t],
                                                        (short)0, acc[t], false, false);
}

// ===========================================================================
// Kernel 1: fused QKV projection.  One wave = 16x64 output tile.
// Q/K written head-major [B,NH,S,D] f16; V written transposed [B,NH,D,S] f16.
// ===========================================================================
__global__ __launch_bounds__(256)
void qkv_proj_kernel(const _Float16* __restrict__ xh, const _Float16* __restrict__ wt,
                     const float* __restrict__ bq, const float* __restrict__ bk,
                     const float* __restrict__ bv,
                     _Float16* __restrict__ hq, _Float16* __restrict__ hk,
                     _Float16* __restrict__ vt) {
    const int z = blockIdx.z;
    const _Float16* A  = xh + (size_t)z * (B_ * S_ * HID_);
    const _Float16* Wt = wt + (size_t)z * (HID_ * HID_);
    const float* bias = (z == 0) ? bq : (z == 1) ? bk : bv;

    const int wave  = blockIdx.x * 8 + (threadIdx.x >> 5);
    const int lane  = threadIdx.x & 31;
    const int lrow  = lane & 15;
    const int kbase = (lane & 16) ? 8 : 0;
    const int m0 = (wave >> 4) * 16;
    const int n0 = (wave & 15) * 64;
    const _Float16* Arow = A + (size_t)(m0 + lrow) * HID_;

    v8f acc[4] = {{}, {}, {}, {}};
    gemm_16x64(Arow,
               Wt + (size_t)(n0 +  0 + lrow) * HID_,
               Wt + (size_t)(n0 + 16 + lrow) * HID_,
               Wt + (size_t)(n0 + 32 + lrow) * HID_,
               Wt + (size_t)(n0 + 48 + lrow) * HID_, kbase, acc);

#pragma unroll
    for (int t = 0; t < 4; ++t) {
        const int col = n0 + 16 * t + lrow;
        const float bn = bias[col];
        const int h = col >> 6, d = col & 63;
#pragma unroll
        for (int i = 0; i < 8; ++i) {
            const int m  = m0 + i + ((lane & 16) ? 8 : 0);
            const int bb = m >> 11, s = m & (S_ - 1);
            const _Float16 val = (_Float16)(acc[t][i] + bn);
            if (z == 0)      hq[(((size_t)bb * NH_ + h) * S_ + s) * D_ + d] = val;
            else if (z == 1) hk[(((size_t)bb * NH_ + h) * S_ + s) * D_ + d] = val;
            else             vt[(((size_t)bb * NH_ + h) * D_ + d) * S_ + s] = val;
        }
    }
}

// ===========================================================================
// Kernel 2: flash attention.  One block = 128 query rows of one (b,h)
// (8 waves x 16 rows).  K/V tiles are async-staged into DOUBLE-BUFFERED LDS
// regions with global_load_async_to_lds_b128: per chunk we wait on our own
// ASYNCcnt, barrier once, immediately issue the next chunk's async loads into
// the other region, then compute QK^T / softmax / P@V on the current region —
// DMA for chunk i+1 overlaps the 8 WMMAs of chunk i.
// ===========================================================================
__global__ __launch_bounds__(256)
void attn_kernel(const _Float16* __restrict__ hq, const _Float16* __restrict__ hk,
                 const _Float16* __restrict__ vt, const float* __restrict__ mask,
                 _Float16* __restrict__ attnbuf) {
    // Single shared object -> LDS offset 0 for async VDST addressing.
    // region r (r=0,1) at halves [r*4096, r*4096+4096):
    //   K tile (32 keys x 64 d) in first 2048 halves, V tile (64 d x 32 keys)
    //   in next 2048 halves.  P staging: 8 x 512 halves at half 8192.
    __shared__ _Float16 smem[2 * 4096 + 8 * 512];    // 24 KB

    const int tid  = threadIdx.x;
    const int wib  = tid >> 5;
    const int lane = tid & 31;
    const int lrow = lane & 15;
    const int kbase = (lane & 16) ? 8 : 0;

    const int bh   = blockIdx.x >> 4;     // (b,h) pair, 16 blocks each
    const int qblk = blockIdx.x & 15;     // 128-query-row slab
    const int b = bh / NH_, h = bh % NH_;

    const _Float16* Q  = hq + (size_t)bh * S_ * D_;
    const _Float16* K  = hk + (size_t)bh * S_ * D_;
    const _Float16* Vt = vt + (size_t)bh * S_ * D_;   // [D][S]
    const float* mrow = mask + (size_t)b * S_;

    const int qt   = qblk * 8 + wib;      // this wave's 16-row q-tile
    const int qrow = qt * 16 + lrow;
    const v16h aq0 = load_frag(Q + (size_t)qrow * D_, kbase);
    const v16h aq1 = load_frag(Q + (size_t)qrow * D_ + 32, kbase);

    // Per-thread source/dest patterns for the async K/V tile staging.
    const int vrow = tid >> 2, vq = tid & 3;          // V: 64 rows x 4 quarters
    const unsigned offK0 = (unsigned)(tid * 16);
    const unsigned offV0 = (unsigned)(4096 + vrow * 64 + vq * 16);

    float mrun[8], lrun[8];
    v8f o0 = {}, o1 = {}, o2 = {}, o3 = {};
#pragma unroll
    for (int i = 0; i < 8; ++i) { mrun[i] = -1e30f; lrun[i] = 0.0f; }
    const float scale = 0.125f;           // 1/sqrt(64)

    // ---- prologue: issue async loads for chunk 0 into region 0 ----
    {
        const _Float16* gk = K  + tid * 8;
        const _Float16* gv = Vt + (size_t)vrow * S_ + vq * 8;
        asm volatile(
            "global_load_async_to_lds_b128 %0, %2, off\n\t"
            "global_load_async_to_lds_b128 %1, %3, off"
            :: "v"(offK0), "v"(offV0),
               "v"((unsigned long long)(uintptr_t)gk),
               "v"((unsigned long long)(uintptr_t)gv)
            : "memory");
    }

    for (int kc = 0; kc < S_; kc += 32) {
        const int cur = (kc >> 5) & 1;
        // Own async loads for region `cur` complete; barrier also guarantees
        // every wave is done READING region cur^1 (previous chunk's compute).
        asm volatile("s_wait_asynccnt 0x0" ::: "memory");
        __syncthreads();

        // ---- issue next chunk's async loads into the other region ----
        if (kc + 32 < S_) {
            const unsigned roff = (unsigned)((cur ^ 1) * 8192);
            const _Float16* gk = K  + (size_t)(kc + 32) * D_ + tid * 8;
            const _Float16* gv = Vt + (size_t)vrow * S_ + (kc + 32) + vq * 8;
            asm volatile(
                "global_load_async_to_lds_b128 %0, %2, off\n\t"
                "global_load_async_to_lds_b128 %1, %3, off"
                :: "v"(offK0 + roff), "v"(offV0 + roff),
                   "v"((unsigned long long)(uintptr_t)gk),
                   "v"((unsigned long long)(uintptr_t)gv)
                : "memory");
        }

        const _Float16* ldsK = smem + cur * 4096;
        const _Float16* ldsV = smem + cur * 4096 + 2048;

        // ---- logits for keys [kc,kc+16) and [kc+16,kc+32) from LDS ----
        v8f c0 = {}, c1 = {};
#pragma unroll
        for (int j = 0; j < 2; ++j) {
            const int kl = 16 * j + lrow;
            const v16h b0 = load_frag(ldsK + kl * 64, kbase);
            const v16h b1 = load_frag(ldsK + kl * 64 + 32, kbase);
            v8f cc = {};
            cc = __builtin_amdgcn_wmma_f32_16x16x32_f16(false, aq0, false, b0,
                                                        (short)0, cc, false, false);
            cc = __builtin_amdgcn_wmma_f32_16x16x32_f16(false, aq1, false, b1,
                                                        (short)0, cc, false, false);
            if (j == 0) c0 = cc; else c1 = cc;
        }

        const float mb0 = (1.0f - mrow[kc + lrow]) * -1e9f;
        const float mb1 = (1.0f - mrow[kc + 16 + lrow]) * -1e9f;

        // ---- online softmax update ----
#pragma unroll
        for (int i = 0; i < 8; ++i) {
            c0[i] = c0[i] * scale + mb0;
            c1[i] = c1[i] * scale + mb1;
            float t = fmaxf(c0[i], c1[i]);
            t = fmaxf(t, __shfl_xor(t, 1, 32));
            t = fmaxf(t, __shfl_xor(t, 2, 32));
            t = fmaxf(t, __shfl_xor(t, 4, 32));
            t = fmaxf(t, __shfl_xor(t, 8, 32));
            const float mnew  = fmaxf(mrun[i], t);
            const float alpha = __expf(mrun[i] - mnew);
            mrun[i] = mnew;
            c0[i] = __expf(c0[i] - mnew);
            c1[i] = __expf(c1[i] - mnew);
            float rs = c0[i] + c1[i];
            rs += __shfl_xor(rs, 1, 32);
            rs += __shfl_xor(rs, 2, 32);
            rs += __shfl_xor(rs, 4, 32);
            rs += __shfl_xor(rs, 8, 32);
            lrun[i] = lrun[i] * alpha + rs;
            o0[i] *= alpha; o1[i] *= alpha; o2[i] *= alpha; o3[i] *= alpha;
        }

        // ---- stage P (C layout) to this wave's LDS slice, reload as A ----
        _Float16* Pl = smem + 8192 + wib * 512;
#pragma unroll
        for (int i = 0; i < 8; ++i) {
            const int r = i + ((lane & 16) ? 8 : 0);
            Pl[r * 32 + lrow]      = (_Float16)c0[i];
            Pl[r * 32 + 16 + lrow] = (_Float16)c1[i];
        }
        asm volatile("s_wait_dscnt 0" ::: "memory");
        const v16h ap = load_frag(Pl + lrow * 32, kbase);

        // ---- O += P @ V over this 32-key chunk (4 d-tiles, V from LDS) ----
#pragma unroll
        for (int t = 0; t < 4; ++t) {
            const int dcol = t * 16 + lrow;
            const v16h bvf = load_frag(ldsV + dcol * 32, kbase);
            v8f a2 = (t == 0) ? o0 : (t == 1) ? o1 : (t == 2) ? o2 : o3;
            a2 = __builtin_amdgcn_wmma_f32_16x16x32_f16(false, ap, false, bvf,
                                                        (short)0, a2, false, false);
            if (t == 0) o0 = a2; else if (t == 1) o1 = a2;
            else if (t == 2) o2 = a2; else o3 = a2;
        }
    }

    // ---- normalize + store; reference's raw reshape (no head transpose):
    //      flat = b*S*HID + h*S*D + s*D + d ----
    const size_t bbase = (size_t)b * S_ * HID_;
#pragma unroll
    for (int i = 0; i < 8; ++i) {
        const float inv = 1.0f / lrun[i];
        const int s = qt * 16 + i + ((lane & 16) ? 8 : 0);
        const size_t f = (size_t)h * S_ * D_ + (size_t)s * D_;
        attnbuf[bbase + f +  0 + lrow] = (_Float16)(o0[i] * inv);
        attnbuf[bbase + f + 16 + lrow] = (_Float16)(o1[i] * inv);
        attnbuf[bbase + f + 32 + lrow] = (_Float16)(o2[i] * inv);
        attnbuf[bbase + f + 48 + lrow] = (_Float16)(o3[i] * inv);
    }
}

// ===========================================================================
// Kernel 3: output projection.  proj = attn @ wd  (bias folded into LN pass)
// ===========================================================================
__global__ __launch_bounds__(256)
void out_proj_kernel(const _Float16* __restrict__ A, const _Float16* __restrict__ Wt,
                     float* __restrict__ Cout) {
    const int wave  = blockIdx.x * 8 + (threadIdx.x >> 5);
    const int lane  = threadIdx.x & 31;
    const int lrow  = lane & 15;
    const int kbase = (lane & 16) ? 8 : 0;
    const int m0 = (wave >> 4) * 16;
    const int n0 = (wave & 15) * 64;
    const _Float16* Arow = A + (size_t)(m0 + lrow) * HID_;

    v8f acc[4] = {{}, {}, {}, {}};
    gemm_16x64(Arow,
               Wt + (size_t)(n0 +  0 + lrow) * HID_,
               Wt + (size_t)(n0 + 16 + lrow) * HID_,
               Wt + (size_t)(n0 + 32 + lrow) * HID_,
               Wt + (size_t)(n0 + 48 + lrow) * HID_, kbase, acc);

#pragma unroll
    for (int t = 0; t < 4; ++t) {
        const int col = n0 + 16 * t + lrow;
#pragma unroll
        for (int i = 0; i < 8; ++i) {
            const int m = m0 + i + ((lane & 16) ? 8 : 0);
            Cout[(size_t)m * HID_ + col] = acc[t][i];
        }
    }
}

// ===========================================================================
// Kernel 4: residual + bias + LayerNorm.  One 256-thread block per row.
// Rows staged into LDS with CDNA5 async global->LDS loads (ASYNCcnt path).
// ===========================================================================
__device__ __forceinline__ float block_sum256(float v, float* red) {
    __syncthreads();
#pragma unroll
    for (int m = 1; m < 32; m <<= 1) v += __shfl_xor(v, m, 32);
    const int wid = threadIdx.x >> 5, lane = threadIdx.x & 31;
    if (lane == 0) red[wid] = v;
    __syncthreads();
    if (threadIdx.x < 32) {
        float r = (lane < 8) ? red[lane] : 0.0f;
#pragma unroll
        for (int m = 1; m < 8; m <<= 1) r += __shfl_xor(r, m, 32);
        if (lane == 0) red[8] = r;
    }
    __syncthreads();
    return red[8];
}

__global__ __launch_bounds__(256)
void ln_kernel(const float* __restrict__ qin, const float* __restrict__ proj,
               const float* __restrict__ bd, const float* __restrict__ gamma,
               const float* __restrict__ beta, float* __restrict__ out) {
    __shared__ float lbuf[2048 + 12];   // [0,1024): q row, [1024,2048): proj row
    const int tid = threadIdx.x;
    const size_t row = blockIdx.x;

    const float* gq = qin  + row * HID_ + tid * 4;
    const float* gp = proj + row * HID_ + tid * 4;
    const unsigned offq = (unsigned)(tid * 16);
    const unsigned offp = (unsigned)(4096 + tid * 16);
    asm volatile(
        "global_load_async_to_lds_b128 %0, %2, off\n\t"
        "global_load_async_to_lds_b128 %1, %3, off\n\t"
        "s_wait_asynccnt 0x0"
        :: "v"(offq), "v"(offp),
           "v"((unsigned long long)(uintptr_t)gq),
           "v"((unsigned long long)(uintptr_t)gp)
        : "memory");

    float xs[4];
    float s = 0.0f;
#pragma unroll
    for (int j = 0; j < 4; ++j) {
        const int e = tid * 4 + j;
        const float x = lbuf[e] + lbuf[1024 + e] + bd[e];
        xs[j] = x;
        s += x;
    }
    const float mu = block_sum256(s, &lbuf[2048]) * (1.0f / HID_);
    float vs = 0.0f;
#pragma unroll
    for (int j = 0; j < 4; ++j) { const float d = xs[j] - mu; vs += d * d; }
    const float var  = block_sum256(vs, &lbuf[2048]) * (1.0f / HID_);
    const float rstd = rsqrtf(var + LN_EPS);
#pragma unroll
    for (int j = 0; j < 4; ++j) {
        const int e = tid * 4 + j;
        out[row * HID_ + e] = gamma[e] * (xs[j] - mu) * rstd + beta[e];
    }
}

// ===========================================================================
extern "C" void kernel_launch(void* const* d_in, const int* in_sizes, int n_in,
                              void* d_out, int out_size, void* d_ws, size_t ws_size,
                              hipStream_t stream) {
    const float* q    = (const float*)d_in[0];
    const float* k    = (const float*)d_in[1];
    const float* v    = (const float*)d_in[2];
    const float* mask = (const float*)d_in[3];
    const float* wq   = (const float*)d_in[4];
    const float* bq   = (const float*)d_in[5];
    const float* wk   = (const float*)d_in[6];
    const float* bk   = (const float*)d_in[7];
    const float* wv   = (const float*)d_in[8];
    const float* bv   = (const float*)d_in[9];
    const float* wd   = (const float*)d_in[10];
    const float* bd   = (const float*)d_in[11];
    const float* gamma= (const float*)d_in[12];
    const float* beta = (const float*)d_in[13];
    float* out = (float*)d_out;

    const size_t nTok = (size_t)B_ * S_ * HID_;    // 4,194,304
    const size_t nW   = (size_t)HID_ * HID_;       // 1,048,576
    char* ws = (char*)d_ws;
    _Float16* xh   = (_Float16*)(ws);                                // 24 MB
    _Float16* wt   = (_Float16*)(ws + 3 * nTok * 2);                 //  8 MB
    _Float16* hq   = (_Float16*)(ws + 3 * nTok * 2 + 4 * nW * 2);    //  8 MB
    _Float16* hk   = (_Float16*)(ws + 4 * nTok * 2 + 4 * nW * 2);    //  8 MB
    _Float16* vtb  = (_Float16*)(ws + 5 * nTok * 2 + 4 * nW * 2);    //  8 MB
    _Float16* attn = (_Float16*)(ws + 6 * nTok * 2 + 4 * nW * 2);    //  8 MB
    float*    proj = (float*)   (ws + 7 * nTok * 2 + 4 * nW * 2);    // 16 MB

    dim3 prepGrid((unsigned)(nTok / 256), 1, 7);
    prep_kernel<<<prepGrid, 256, 0, stream>>>(q, k, v, wq, wk, wv, wd, xh, wt);

    const int waves = (B_ * S_ / 16) * (HID_ / 64);    // 4096 waves / GEMM
    dim3 qkvGrid(waves / 8, 1, 3);
    qkv_proj_kernel<<<qkvGrid, 256, 0, stream>>>(xh, wt, bq, bk, bv, hq, hk, vtb);
    attn_kernel<<<32 * 16, 256, 0, stream>>>(hq, hk, vtb, mask, attn);
    out_proj_kernel<<<waves / 8, 256, 0, stream>>>(attn, wt + 3 * nW, proj);
    ln_kernel<<<B_ * S_, 256, 0, stream>>>(q, proj, bd, gamma, beta, out);
}
